// STGNN_30777735644031
// MI455X (gfx1250) — matrix operand order repaired and checked by reading, not compiled, not run on hardware
//
#include <hip/hip_runtime.h>
#include <math.h>

typedef _Float16 half_t;
typedef __attribute__((ext_vector_type(16))) _Float16 v16h;
typedef __attribute__((ext_vector_type(8)))  _Float16 v8h;
typedef __attribute__((ext_vector_type(8)))  float    v8f;

constexpr int  cB   = 32, cT = 64, cN = 512, cF = 32, cH = 64;
constexpr int  cBT  = cB * cT;           // 2048
constexpr int  cBN  = cB * cN;           // 16384
constexpr long cBTN = (long)cBT * cN;    // 1,048,576
constexpr int  cG   = 4 * cH;            // 256 gate columns
constexpr int  cHP  = cH + 8;            // padded LDS h row (bank spread, 16B-aligned rows)

// ---------------- WMMA fragment helpers (CDNA5 16x16x32 f16 layouts) ----------------
// A 16x32 (MxK), row-major source (k contiguous): per-lane two 16B runs at +0 / +32B
static __device__ __forceinline__ v16h load_a_frag(const half_t* __restrict__ A,
                                                   long lda, long m0, int k0) {
  const int lane = threadIdx.x & 31;
  const int r  = lane & 15;
  const int kh = lane >> 4;
  const half_t* p = A + (m0 + r) * lda + k0 + 8 * kh;
  v8h lo = *(const v8h*)(p);
  v8h hi = *(const v8h*)(p + 16);
  v16h a;
#pragma unroll
  for (int i = 0; i < 8; ++i) { a[i] = lo[i]; a[i + 8] = hi[i]; }
  return a;
}

// Same A fragment but from LDS (generic pointer into __shared__; emits ds_load_b128)
static __device__ __forceinline__ v16h lds_a_frag(const half_t* Hs, int ldh, int m0, int k0) {
  const int lane = threadIdx.x & 31;
  const int r  = lane & 15;
  const int kh = lane >> 4;
  const half_t* p = Hs + (m0 + r) * ldh + k0 + 8 * kh;
  v8h lo = *(const v8h*)(p);
  v8h hi = *(const v8h*)(p + 16);
  v16h a;
#pragma unroll
  for (int i = 0; i < 8; ++i) { a[i] = lo[i]; a[i + 8] = hi[i]; }
  return a;
}

// B 32x16 (KxN) fragment read from B^T storage Bt[n][k] (k contiguous): per-lane two 16B runs at +0 / +32B
static __device__ __forceinline__ v16h load_bt_frag(const half_t* __restrict__ Bt,
                                                    long ldb, int k0, int n0) {
  const int lane = threadIdx.x & 31;
  const int c  = lane & 15;
  const int kh = lane >> 4;
  const half_t* p = Bt + (long)(n0 + c) * ldb + k0 + 8 * kh;
  v8h lo = *(const v8h*)(p);   // 16B-aligned at every call site
  v8h hi = *(const v8h*)(p + 16);
  v16h b;
#pragma unroll
  for (int i = 0; i < 8; ++i) { b[i] = lo[i]; b[i + 8] = hi[i]; }
  return b;
}

static __device__ __forceinline__ v8f wmma_f16(v16h a, v16h b, v8f c) {
  return __builtin_amdgcn_wmma_f32_16x16x32_f16(false, a, false, b, (short)0, c,
                                                false, false);
}

// C/D tile: lane holds n = lane&15, m = v + 8*(lane>>4)
static __device__ __forceinline__ void store_tile_f16(half_t* __restrict__ C, long ldc,
                                                      long m0, int n0, v8f acc) {
  const int lane = threadIdx.x & 31, c = lane & 15, kh = lane >> 4;
#pragma unroll
  for (int v = 0; v < 8; ++v)
    C[(m0 + v + 8 * kh) * ldc + n0 + c] = (half_t)acc[v];
}

static __device__ __forceinline__ float sigmoidf_(float x) {
  return 1.0f / (1.0f + expf(-x));
}

// ---------------- prep kernels ----------------
__global__ void k_cast_f16(const float* __restrict__ a, half_t* __restrict__ o, int n) {
  int i = blockIdx.x * 256 + threadIdx.x;
  if (i < n) o[i] = (half_t)a[i];
}

__global__ void k_bsum(const float* __restrict__ a, const float* __restrict__ b,
                       float* __restrict__ o, int n) {
  int i = blockIdx.x * 256 + threadIdx.x;
  if (i < n) o[i] = a[i] + b[i];
}

// x (bt, n=512, f=32) f32  ->  xT (bt, f=32, n=512) f16   (LDS-tiled 32x32)
__global__ __launch_bounds__(256) void k_xT(const float* __restrict__ x,
                                            half_t* __restrict__ xT) {
  __shared__ half_t t[32][33];
  const int tx = threadIdx.x, ty = threadIdx.y;   // block (32,8)
  const int bt = blockIdx.z;
  const int nb = blockIdx.y * 32;
  const float* s = x + (long)bt * cN * cF;
  half_t* d = xT + (long)bt * cF * cN;
#pragma unroll
  for (int j = 0; j < 4; ++j)
    t[ty + 8 * j][tx] = (half_t)s[(long)(nb + ty + 8 * j) * cF + tx];
  __syncthreads();
#pragma unroll
  for (int j = 0; j < 4; ++j)
    d[(long)(ty + 8 * j) * cN + nb + tx] = t[tx][ty + 8 * j];
}

// ------- stage 1: agg0[bt](512x32) = adj(512x512) x x[bt];  B^T = xT (f,node) -------
__global__ __launch_bounds__(128) void k_agg0(const half_t* __restrict__ adj16,
                                              const half_t* __restrict__ xT,
                                              half_t* __restrict__ out) {
  const int wave = threadIdx.x >> 5;
  const int m0 = (blockIdx.x * 4 + wave) * 32;
  const int bt = blockIdx.z;
  const half_t* Bt = xT + (long)bt * cF * cN;
  half_t* Cm = out + (long)bt * cN * cF;
  v8f a00 = {}, a01 = {}, a10 = {}, a11 = {};
  for (int k0 = 0; k0 < cN; k0 += 32) {
    v16h fa0 = load_a_frag(adj16, cN, m0, k0);
    v16h fa1 = load_a_frag(adj16, cN, m0 + 16, k0);
    v16h fb0 = load_bt_frag(Bt, cN, k0, 0);
    v16h fb1 = load_bt_frag(Bt, cN, k0, 16);
    a00 = wmma_f16(fa0, fb0, a00);
    a01 = wmma_f16(fa0, fb1, a01);
    a10 = wmma_f16(fa1, fb0, a10);
    a11 = wmma_f16(fa1, fb1, a11);
  }
  store_tile_f16(Cm, cF, m0,      0,  a00);
  store_tile_f16(Cm, cF, m0,      16, a01);
  store_tile_f16(Cm, cF, m0 + 16, 0,  a10);
  store_tile_f16(Cm, cF, m0 + 16, 16, a11);
}

// ------- stage 2 (transposed): h0T[bt](64x512) = relu(Wg0(64x32) x agg0[bt]^T + b) -------
__global__ __launch_bounds__(128) void k_lin0T(const half_t* __restrict__ agg0,
                                               const half_t* __restrict__ Wg0,
                                               const float* __restrict__ bias,
                                               half_t* __restrict__ h0T) {
  const int wave = threadIdx.x >> 5;
  const int n0 = (blockIdx.x * 4 + wave) * 32;  // node strip
  const int m0 = blockIdx.y * 32;               // h strip
  const int bt = blockIdx.z;
  const half_t* Bt = agg0 + (long)bt * cN * cF;
  half_t* Cm = h0T + (long)bt * cH * cN;
  const int lane = threadIdx.x & 31, kh = lane >> 4;
  v8f acc[2][2];
#pragma unroll
  for (int mi = 0; mi < 2; ++mi)
#pragma unroll
    for (int v = 0; v < 8; ++v) {
      float bz = bias[m0 + mi * 16 + v + 8 * kh];   // bias per output row (h)
      acc[mi][0][v] = bz;
      acc[mi][1][v] = bz;
    }
  v16h fa0 = load_a_frag(Wg0, cF, m0, 0);
  v16h fa1 = load_a_frag(Wg0, cF, m0 + 16, 0);
  v16h fb0 = load_bt_frag(Bt, cF, 0, n0);
  v16h fb1 = load_bt_frag(Bt, cF, 0, n0 + 16);
  acc[0][0] = wmma_f16(fa0, fb0, acc[0][0]);
  acc[0][1] = wmma_f16(fa0, fb1, acc[0][1]);
  acc[1][0] = wmma_f16(fa1, fb0, acc[1][0]);
  acc[1][1] = wmma_f16(fa1, fb1, acc[1][1]);
#pragma unroll
  for (int mi = 0; mi < 2; ++mi)
#pragma unroll
    for (int ni = 0; ni < 2; ++ni) {
      v8f r;
#pragma unroll
      for (int v = 0; v < 8; ++v) r[v] = acc[mi][ni][v] > 0.0f ? acc[mi][ni][v] : 0.0f;
      store_tile_f16(Cm, cN, m0 + mi * 16, n0 + ni * 16, r);
    }
}

// ------- stage 3: agg1[bt](512x64) = adj x h0[bt];  B^T = h0T (h,node) -------
__global__ __launch_bounds__(128) void k_agg1(const half_t* __restrict__ adj16,
                                              const half_t* __restrict__ h0T,
                                              half_t* __restrict__ out) {
  const int wave = threadIdx.x >> 5;
  const int m0 = (blockIdx.x * 4 + wave) * 32;
  const int n0 = blockIdx.y * 32;
  const int bt = blockIdx.z;
  const half_t* Bt = h0T + (long)bt * cH * cN;
  half_t* Cm = out + (long)bt * cN * cH;
  v8f a00 = {}, a01 = {}, a10 = {}, a11 = {};
  for (int k0 = 0; k0 < cN; k0 += 32) {
    v16h fa0 = load_a_frag(adj16, cN, m0, k0);
    v16h fa1 = load_a_frag(adj16, cN, m0 + 16, k0);
    v16h fb0 = load_bt_frag(Bt, cN, k0, n0);
    v16h fb1 = load_bt_frag(Bt, cN, k0, n0 + 16);
    a00 = wmma_f16(fa0, fb0, a00);
    a01 = wmma_f16(fa0, fb1, a01);
    a10 = wmma_f16(fa1, fb0, a10);
    a11 = wmma_f16(fa1, fb1, a11);
  }
  store_tile_f16(Cm, cH, m0,      n0,      a00);
  store_tile_f16(Cm, cH, m0,      n0 + 16, a01);
  store_tile_f16(Cm, cH, m0 + 16, n0,      a10);
  store_tile_f16(Cm, cH, m0 + 16, n0 + 16, a11);
}

// ------- stage 4: h1 = agg1 @ Wg1^T + b  ->  seq (b,n,t,H) layout -------
__global__ __launch_bounds__(128) void k_lin1(const half_t* __restrict__ A,   // cBTN x H
                                              const half_t* __restrict__ Wg1, // 64x64 (B^T)
                                              const float* __restrict__ bias,
                                              half_t* __restrict__ seq) {
  const int wave = threadIdx.x >> 5;
  const long m0 = ((long)blockIdx.x * 4 + wave) * 32;
  const int n0 = blockIdx.y * 32;
  const int lane = threadIdx.x & 31, c = lane & 15, kh = lane >> 4;
  v8f acc[2][2];
#pragma unroll
  for (int ni = 0; ni < 2; ++ni) {
    float bz = bias[n0 + ni * 16 + c];
#pragma unroll
    for (int v = 0; v < 8; ++v) { acc[0][ni][v] = bz; acc[1][ni][v] = bz; }
  }
#pragma unroll
  for (int k0 = 0; k0 < cH; k0 += 32) {
    v16h fa0 = load_a_frag(A, cH, m0, k0);
    v16h fa1 = load_a_frag(A, cH, m0 + 16, k0);
    v16h fb0 = load_bt_frag(Wg1, cH, k0, n0);
    v16h fb1 = load_bt_frag(Wg1, cH, k0, n0 + 16);
    acc[0][0] = wmma_f16(fa0, fb0, acc[0][0]);
    acc[0][1] = wmma_f16(fa0, fb1, acc[0][1]);
    acc[1][0] = wmma_f16(fa1, fb0, acc[1][0]);
    acc[1][1] = wmma_f16(fa1, fb1, acc[1][1]);
  }
  const int bt = (int)(m0 / cN);          // all 32 rows in one (b,t) slice
  const int node_base = (int)(m0 % cN);
  const int t = bt % cT, b = bt / cT;
#pragma unroll
  for (int mi = 0; mi < 2; ++mi)
#pragma unroll
    for (int v = 0; v < 8; ++v) {
      int node = node_base + mi * 16 + v + 8 * kh;
      long rowoff = (((long)b * cN + node) * cT + t) * cH;
      seq[rowoff + n0 + c]      = (half_t)acc[mi][0][v];
      seq[rowoff + n0 + 16 + c] = (half_t)acc[mi][1][v];
    }
}

// ------- persistent 2-layer LSTM: one kernel for all 64 timesteps -------
// block = 32 rows; h0/h1 in LDS, c0/c1 in registers; A-frags for recurrent GEMMs via ds_load
__global__ __launch_bounds__(256) void k_lstm_all(const half_t* __restrict__ seq,   // (bn,t,H)
                                                  const half_t* __restrict__ Wih0,  // 256x64 B^T
                                                  const half_t* __restrict__ Whh0,
                                                  const float*  __restrict__ bs0,
                                                  const half_t* __restrict__ Wih1,
                                                  const half_t* __restrict__ Whh1,
                                                  const float*  __restrict__ bs1,
                                                  half_t* __restrict__ h1out) {     // BN x 64
  __shared__ float gates[32][cG];
  __shared__ __align__(32) half_t h0s[32][cHP];
  __shared__ __align__(32) half_t h1s[32][cHP];
  const int tid = threadIdx.x;
  const int wave = tid >> 5;
  const long rows0 = (long)blockIdx.x * 32;
  const int n0 = wave * 32;
  const int lane = tid & 31, c = lane & 15, kh = lane >> 4;

  for (int i = tid; i < 32 * cHP; i += 256) {
    (&h0s[0][0])[i] = (half_t)0.0f;
    (&h1s[0][0])[i] = (half_t)0.0f;
  }
  float c0r[8], c1r[8];
#pragma unroll
  for (int i = 0; i < 8; ++i) { c0r[i] = 0.0f; c1r[i] = 0.0f; }
  const float bz0a = bs0[n0 + c], bz0b = bs0[n0 + 16 + c];
  const float bz1a = bs1[n0 + c], bz1b = bs1[n0 + 16 + c];
  __syncthreads();

#pragma unroll 1
  for (int t = 0; t < cT; ++t) {
    // ---- layer 0: gates0 = x_t @ Wih0^T + h0 @ Whh0^T + b ----
    v8f acc[2][2];
#pragma unroll
    for (int v = 0; v < 8; ++v) {
      acc[0][0][v] = bz0a; acc[1][0][v] = bz0a;
      acc[0][1][v] = bz0b; acc[1][1][v] = bz0b;
    }
    const half_t* X = seq + (long)t * cH;
#pragma unroll
    for (int k0 = 0; k0 < cH; k0 += 32) {
      v16h fa0 = load_a_frag(X, (long)cT * cH, rows0, k0);
      v16h fa1 = load_a_frag(X, (long)cT * cH, rows0 + 16, k0);
      v16h fb0 = load_bt_frag(Wih0, cH, k0, n0);
      v16h fb1 = load_bt_frag(Wih0, cH, k0, n0 + 16);
      acc[0][0] = wmma_f16(fa0, fb0, acc[0][0]);
      acc[0][1] = wmma_f16(fa0, fb1, acc[0][1]);
      acc[1][0] = wmma_f16(fa1, fb0, acc[1][0]);
      acc[1][1] = wmma_f16(fa1, fb1, acc[1][1]);
    }
#pragma unroll
    for (int k0 = 0; k0 < cH; k0 += 32) {
      v16h fa0 = lds_a_frag(&h0s[0][0], cHP, 0, k0);
      v16h fa1 = lds_a_frag(&h0s[0][0], cHP, 16, k0);
      v16h fb0 = load_bt_frag(Whh0, cH, k0, n0);
      v16h fb1 = load_bt_frag(Whh0, cH, k0, n0 + 16);
      acc[0][0] = wmma_f16(fa0, fb0, acc[0][0]);
      acc[0][1] = wmma_f16(fa0, fb1, acc[0][1]);
      acc[1][0] = wmma_f16(fa1, fb0, acc[1][0]);
      acc[1][1] = wmma_f16(fa1, fb1, acc[1][1]);
    }
#pragma unroll
    for (int mi = 0; mi < 2; ++mi)
#pragma unroll
      for (int ni = 0; ni < 2; ++ni)
#pragma unroll
        for (int v = 0; v < 8; ++v)
          gates[mi * 16 + v + 8 * kh][n0 + ni * 16 + c] = acc[mi][ni][v];
    __syncthreads();
#pragma unroll
    for (int rep = 0; rep < 8; ++rep) {
      int q = rep * 256 + tid;               // 32 rows x 64 h-cols
      int r = q >> 6, cc = q & 63;
      float ig = sigmoidf_(gates[r][cc]);
      float fg = sigmoidf_(gates[r][cc + cH]);
      float gg = tanhf(gates[r][cc + 2 * cH]);
      float og = sigmoidf_(gates[r][cc + 3 * cH]);
      float cn = fg * c0r[rep] + ig * gg;
      c0r[rep] = cn;
      h0s[r][cc] = (half_t)(og * tanhf(cn));
    }
    __syncthreads();   // h0 new visible; gates0 fully consumed

    // ---- layer 1: gates1 = h0_new @ Wih1^T + h1 @ Whh1^T + b ----
#pragma unroll
    for (int v = 0; v < 8; ++v) {
      acc[0][0][v] = bz1a; acc[1][0][v] = bz1a;
      acc[0][1][v] = bz1b; acc[1][1][v] = bz1b;
    }
#pragma unroll
    for (int k0 = 0; k0 < cH; k0 += 32) {
      v16h fa0 = lds_a_frag(&h0s[0][0], cHP, 0, k0);
      v16h fa1 = lds_a_frag(&h0s[0][0], cHP, 16, k0);
      v16h fb0 = load_bt_frag(Wih1, cH, k0, n0);
      v16h fb1 = load_bt_frag(Wih1, cH, k0, n0 + 16);
      acc[0][0] = wmma_f16(fa0, fb0, acc[0][0]);
      acc[0][1] = wmma_f16(fa0, fb1, acc[0][1]);
      acc[1][0] = wmma_f16(fa1, fb0, acc[1][0]);
      acc[1][1] = wmma_f16(fa1, fb1, acc[1][1]);
    }
#pragma unroll
    for (int k0 = 0; k0 < cH; k0 += 32) {
      v16h fa0 = lds_a_frag(&h1s[0][0], cHP, 0, k0);
      v16h fa1 = lds_a_frag(&h1s[0][0], cHP, 16, k0);
      v16h fb0 = load_bt_frag(Whh1, cH, k0, n0);
      v16h fb1 = load_bt_frag(Whh1, cH, k0, n0 + 16);
      acc[0][0] = wmma_f16(fa0, fb0, acc[0][0]);
      acc[0][1] = wmma_f16(fa0, fb1, acc[0][1]);
      acc[1][0] = wmma_f16(fa1, fb0, acc[1][0]);
      acc[1][1] = wmma_f16(fa1, fb1, acc[1][1]);
    }
#pragma unroll
    for (int mi = 0; mi < 2; ++mi)
#pragma unroll
      for (int ni = 0; ni < 2; ++ni)
#pragma unroll
        for (int v = 0; v < 8; ++v)
          gates[mi * 16 + v + 8 * kh][n0 + ni * 16 + c] = acc[mi][ni][v];
    __syncthreads();
#pragma unroll
    for (int rep = 0; rep < 8; ++rep) {
      int q = rep * 256 + tid;
      int r = q >> 6, cc = q & 63;
      float ig = sigmoidf_(gates[r][cc]);
      float fg = sigmoidf_(gates[r][cc + cH]);
      float gg = tanhf(gates[r][cc + 2 * cH]);
      float og = sigmoidf_(gates[r][cc + 3 * cH]);
      float cn = fg * c1r[rep] + ig * gg;
      c1r[rep] = cn;
      float hn = og * tanhf(cn);
      h1s[r][cc] = (half_t)hn;
      if (t == cT - 1) h1out[(rows0 + r) * cH + cc] = (half_t)hn;
    }
    __syncthreads();   // h1 new visible; gates1 fully consumed before next step
  }
}

// ------- output projection: out = lastH @ Wout^T + b_out (f32) -------
__global__ __launch_bounds__(128) void k_out(const half_t* __restrict__ Hl,   // BN x 64
                                             const half_t* __restrict__ Wout, // 32x64 (B^T)
                                             const float* __restrict__ bias,
                                             float* __restrict__ out) {       // BN x 32
  const int wave = threadIdx.x >> 5;
  const long m0 = ((long)blockIdx.x * 4 + wave) * 32;
  const int lane = threadIdx.x & 31, c = lane & 15, kh = lane >> 4;
  v8f acc[2][2];
#pragma unroll
  for (int ni = 0; ni < 2; ++ni) {
    float bz = bias[ni * 16 + c];
#pragma unroll
    for (int v = 0; v < 8; ++v) { acc[0][ni][v] = bz; acc[1][ni][v] = bz; }
  }
#pragma unroll
  for (int k0 = 0; k0 < cH; k0 += 32) {
    v16h fa0 = load_a_frag(Hl, cH, m0, k0);
    v16h fa1 = load_a_frag(Hl, cH, m0 + 16, k0);
    v16h fb0 = load_bt_frag(Wout, cH, k0, 0);
    v16h fb1 = load_bt_frag(Wout, cH, k0, 16);
    acc[0][0] = wmma_f16(fa0, fb0, acc[0][0]);
    acc[0][1] = wmma_f16(fa0, fb1, acc[0][1]);
    acc[1][0] = wmma_f16(fa1, fb0, acc[1][0]);
    acc[1][1] = wmma_f16(fa1, fb1, acc[1][1]);
  }
#pragma unroll
  for (int mi = 0; mi < 2; ++mi)
#pragma unroll
    for (int ni = 0; ni < 2; ++ni)
#pragma unroll
      for (int v = 0; v < 8; ++v)
        out[(m0 + mi * 16 + v + 8 * kh) * cF + ni * 16 + c] = acc[mi][ni][v];
}

extern "C" void kernel_launch(void* const* d_in, const int* in_sizes, int n_in,
                              void* d_out, int out_size, void* d_ws, size_t ws_size,
                              hipStream_t stream) {
  (void)in_sizes; (void)n_in; (void)out_size; (void)ws_size;
  const float* x     = (const float*)d_in[0];
  const float* adj   = (const float*)d_in[1];
  const float* W_g0  = (const float*)d_in[2];
  const float* b_g0  = (const float*)d_in[3];
  const float* W_g1  = (const float*)d_in[4];
  const float* b_g1  = (const float*)d_in[5];
  const float* W_ih0 = (const float*)d_in[6];
  const float* W_hh0 = (const float*)d_in[7];
  const float* b_ih0 = (const float*)d_in[8];
  const float* b_hh0 = (const float*)d_in[9];
  const float* W_ih1 = (const float*)d_in[10];
  const float* W_hh1 = (const float*)d_in[11];
  const float* b_ih1 = (const float*)d_in[12];
  const float* b_hh1 = (const float*)d_in[13];
  const float* W_out = (const float*)d_in[14];
  const float* b_out = (const float*)d_in[15];

  // ---- workspace carve (~260 MB), 256B aligned ----
  char* w = (char*)d_ws;
  size_t off = 0;
  auto take = [&](size_t bytes) -> char* {
    char* p = w + off;
    off = (off + bytes + 255) & ~(size_t)255;
    return p;
  };
  half_t* adj16 = (half_t*)take((size_t)cN * cN * 2);
  half_t* wg0c  = (half_t*)take((size_t)cH * cF * 2);
  half_t* wg1c  = (half_t*)take((size_t)cH * cH * 2);
  half_t* wih0c = (half_t*)take((size_t)cG * cH * 2);
  half_t* whh0c = (half_t*)take((size_t)cG * cH * 2);
  half_t* wih1c = (half_t*)take((size_t)cG * cH * 2);
  half_t* whh1c = (half_t*)take((size_t)cG * cH * 2);
  half_t* woutc = (half_t*)take((size_t)cF * cH * 2);
  float*  bsum0 = (float*)take((size_t)cG * 4);
  float*  bsum1 = (float*)take((size_t)cG * 4);
  half_t* h1buf = (half_t*)take((size_t)cBN * cH * 2);
  half_t* R1 = (half_t*)take((size_t)cBTN * cH * 2);  // agg0 (64MB), later agg1 (128MB)
  half_t* RT = (half_t*)take((size_t)cBTN * cH * 2);  // xT (64MB) -> h0T -> seq

  // ---- prep: f16 casts (no transposes), bias sums, xT ----
  k_cast_f16<<<(cN * cN + 255) / 256, 256, 0, stream>>>(adj, adj16, cN * cN);
  k_cast_f16<<<(cH * cF + 255) / 256, 256, 0, stream>>>(W_g0, wg0c, cH * cF);
  k_cast_f16<<<(cH * cH + 255) / 256, 256, 0, stream>>>(W_g1, wg1c, cH * cH);
  k_cast_f16<<<(cG * cH + 255) / 256, 256, 0, stream>>>(W_ih0, wih0c, cG * cH);
  k_cast_f16<<<(cG * cH + 255) / 256, 256, 0, stream>>>(W_hh0, whh0c, cG * cH);
  k_cast_f16<<<(cG * cH + 255) / 256, 256, 0, stream>>>(W_ih1, wih1c, cG * cH);
  k_cast_f16<<<(cG * cH + 255) / 256, 256, 0, stream>>>(W_hh1, whh1c, cG * cH);
  k_cast_f16<<<(cF * cH + 255) / 256, 256, 0, stream>>>(W_out, woutc, cF * cH);
  k_bsum<<<1, 256, 0, stream>>>(b_ih0, b_hh0, bsum0, cG);
  k_bsum<<<1, 256, 0, stream>>>(b_ih1, b_hh1, bsum1, cG);
  k_xT<<<dim3(1, cN / 32, cBT), dim3(32, 8), 0, stream>>>(x, RT);

  // ---- GCN pipeline (ping-pong R1/RT) ----
  k_agg0 <<<dim3(4, 1, cBT), 128, 0, stream>>>(adj16, RT, R1);
  k_lin0T<<<dim3(4, 2, cBT), 128, 0, stream>>>(R1, wg0c, b_g0, RT);
  k_agg1 <<<dim3(4, 2, cBT), 128, 0, stream>>>(adj16, RT, R1);
  k_lin1 <<<dim3((int)(cBTN / 128), 2, 1), 128, 0, stream>>>(R1, wg1c, b_g1, RT);

  // ---- persistent 2-layer LSTM over all timesteps (single launch) ----
  k_lstm_all<<<cBN / 32, 256, 0, stream>>>(RT, wih0c, whh0c, bsum0,
                                           wih1c, whh1c, bsum1, h1buf);

  // ---- output projection ----
  k_out<<<dim3(cBN / 128, 1), 128, 0, stream>>>(h1buf, woutc, b_out, (float*)d_out);
}
